// HypersphericalPrototypeBank_63367947485821
// MI455X (gfx1250) — matrix operand order repaired and checked by reading, not compiled, not run on hardware
//
#include <hip/hip_runtime.h>
#include <hip/hip_bf16.h>

// ---- problem dimensions (fixed by the reference) ----
#define BR   4
#define BB   32
#define PP   576
#define TT   577
#define DD   768
#define KK   2048
#define NN   (BB * PP)     // 18432 tokens per batch-replica
#define NTOK (BR * NN)     // 73728 tokens total

typedef __attribute__((ext_vector_type(16))) __bf16 v16bf;
typedef __attribute__((ext_vector_type(8)))  float  v8f;
typedef __attribute__((ext_vector_type(4)))  unsigned int v4u;

__device__ __forceinline__ float waveSum(float v) {
#pragma unroll
    for (int off = 16; off >= 1; off >>= 1) v += __shfl_xor(v, off, 32);
    return v;
}

// ---------------------------------------------------------------------------
// Kernel 0: zero scratch (sums + counts, contiguous in ws)
// ---------------------------------------------------------------------------
__global__ void k_zero(float* __restrict__ p, size_t n) {
    size_t i = (size_t)blockIdx.x * blockDim.x + threadIdx.x;
    size_t stride = (size_t)gridDim.x * blockDim.x;
    for (; i < n; i += stride) p[i] = 0.0f;
}

// ---------------------------------------------------------------------------
// Kernel 1: drop CLS, L2-normalize each patch token, emit bf16 row-major.
// One wave32 per token; 768/32 = 24 elements per lane.
// ---------------------------------------------------------------------------
__global__ void __launch_bounds__(256) k_norm_tokens(const float* __restrict__ img,
                                                     __bf16* __restrict__ toksB) {
    int wave = (int)((blockIdx.x * blockDim.x + threadIdx.x) >> 5);
    int lane = threadIdx.x & 31;
    if (wave >= NTOK) return;
    int br = wave / NN;
    int n  = wave % NN;
    int b = n / PP, p = n % PP;
    const float* src = img + (((size_t)(br * BB + b)) * TT + (1 + p)) * DD;
    float x[24];
    float ss = 0.0f;
#pragma unroll
    for (int i = 0; i < 24; ++i) { x[i] = src[lane + 32 * i]; ss += x[i] * x[i]; }
    ss = waveSum(ss);
    float inv = 1.0f / fmaxf(sqrtf(ss), 1e-12f);
    __bf16* dst = toksB + (size_t)wave * DD;
#pragma unroll
    for (int i = 0; i < 24; ++i) dst[lane + 32 * i] = (__bf16)(x[i] * inv);
}

// ---------------------------------------------------------------------------
// Kernel 1b: prototypes f32 -> bf16 (already unit-norm)
// ---------------------------------------------------------------------------
__global__ void k_cvt_protos(const float* __restrict__ pro, __bf16* __restrict__ proB) {
    size_t total = (size_t)BR * KK * DD;
    size_t i = (size_t)blockIdx.x * blockDim.x + threadIdx.x;
    size_t stride = (size_t)gridDim.x * blockDim.x;
    for (; i < total; i += stride) proB[i] = (__bf16)pro[i];
}

// ---------------------------------------------------------------------------
// Kernel 2: fused cosine-sim GEMM + argmax via v_wmma_f32_16x16x32_bf16.
//
// Workgroup = 8 waves = 8 adjacent 16-token M-tiles of one batch replica.
// All waves share 16-proto B tiles staged in LDS by the gfx1250 async
// global->LDS engine (global_load_async_to_lds_b128, ASYNCcnt), double-
// buffered so the next tile's fetch overlaps the current WMMA chain.
//
// Fragment layouts per the CDNA5 ISA:
//  A (16x32 bf16, MxK): lane (m = lane&15, h = lane>>4):
//      a[0..7]  = K[8h .. 8h+7], a[8..15] = K[16+8h .. 16+8h+7]  (2x b128)
//  B (32x16 bf16, KxN): lane (n = lane&15, h): b[0..7] = K[8h..8h+7], b[8..15] = K[16+8h..16+8h+7]
//      -> 2x ds_load_b128 from proto row n in LDS
//  C (16x16 f32): VGPR r -> row M = r + 8h, column N = lane&15.
// ---------------------------------------------------------------------------
#define TILE_BYTES (16 * DD * 2)   // 24576 B per B tile

__global__ void __launch_bounds__(256) k_argmax_wmma(const __bf16* __restrict__ toksB,
                                                     const __bf16* __restrict__ proB,
                                                     int* __restrict__ assign) {
    extern __shared__ __bf16 smem[];                 // 2 * TILE_BYTES = 49152 B
    const int tilesPerBr = NN / 16;                  // 1152
    const int wgPerBr    = tilesPerBr / 8;           // 144
    int br       = blockIdx.x / wgPerBr;
    int tileBase = (blockIdx.x % wgPerBr) * 8;
    int w        = threadIdx.x >> 5;
    int lane     = threadIdx.x & 31;
    int token0   = (tileBase + w) * 16;
    int m = lane & 15;
    int h = lane >> 4;

    union Frag { v16bf bf; v4u u[2]; };

    // Preload this wave's full A tile into registers (24 frags = 192 VGPRs).
    const __bf16* arow = toksB + ((size_t)br * NN + token0 + m) * DD;
    Frag A[24];
#pragma unroll
    for (int dc = 0; dc < 24; ++dc) {
        A[dc].u[0] = *(const v4u*)(arow + dc * 32 + 8 * h);
        A[dc].u[1] = *(const v4u*)(arow + dc * 32 + 16 + 8 * h);
    }

    // Async-stage one 16-proto B tile (row-major bf16) into LDS buffer buf.
    const char* proBase = (const char*)(proB + (size_t)br * KK * DD);
    auto stage = [&](int pt, int buf) {
        const char* src = proBase + (size_t)pt * TILE_BYTES;
        unsigned ldsBase = (unsigned)(buf * TILE_BYTES);
        int t = (int)threadIdx.x;
#pragma unroll
        for (int i = 0; i < 6; ++i) {                // 1536 x 16B chunks / 256 thr
            unsigned off = (unsigned)(t + i * 256) * 16u;
            const char* g = src + off;
            unsigned    l = ldsBase + off;
            asm volatile("global_load_async_to_lds_b128 %0, %1, off"
                         :: "v"(l), "v"(g) : "memory");
        }
    };

    float bestv[8];
    int   besti[8];
#pragma unroll
    for (int r = 0; r < 8; ++r) { bestv[r] = -3.402823e38f; besti[r] = 0; }

    stage(0, 0);
    for (int pt = 0; pt < KK / 16; ++pt) {
        // my async copies for the current buffer have landed in LDS...
        asm volatile("s_wait_asynccnt 0x0" ::: "memory");
        // ...and everyone is done reading the buffer we are about to refill
        __syncthreads();
        if (pt + 1 < KK / 16) stage(pt + 1, (pt + 1) & 1);

        const __bf16* bbase = smem + (size_t)(pt & 1) * (16 * DD);
        const __bf16* brow  = bbase + m * DD + 8 * h;
        v8f c = {};
#pragma unroll
        for (int dc = 0; dc < 24; ++dc) {
            Frag B;
            B.u[0] = *(const v4u*)(brow + dc * 32);
            B.u[1] = *(const v4u*)(brow + dc * 32 + 16);
            c = __builtin_amdgcn_wmma_f32_16x16x32_bf16(
                    false, A[dc].bf, false, B.bf, (short)0, c, false, false);
        }

        // per-row argmax over the 16 columns of this tile (butterfly within
        // each 16-lane half; halves hold disjoint rows)
#pragma unroll
        for (int r = 0; r < 8; ++r) {
            float v = c[r];
            int col = m;
#pragma unroll
            for (int off = 1; off < 16; off <<= 1) {
                float ov = __shfl_xor(v, off, 32);
                int   oc = __shfl_xor(col, off, 32);
                if (ov > v || (ov == v && oc < col)) { v = ov; col = oc; }
            }
            int kidx = pt * 16 + col;
            if (v > bestv[r]) { bestv[r] = v; besti[r] = kidx; }
        }
    }

    if (m == 0) {
#pragma unroll
        for (int r = 0; r < 8; ++r) {
            int row = r + 8 * h;                     // C layout: M = r + 8h
            assign[(size_t)br * NN + token0 + row] = besti[r];
        }
    }
}

// ---------------------------------------------------------------------------
// Kernel 3: masked scatter-add of normalized tokens into per-prototype sums.
// One wave per token; recompute normalization from original f32 tokens.
// ---------------------------------------------------------------------------
__global__ void __launch_bounds__(256) k_scatter(const float* __restrict__ img,
                                                 const unsigned char* __restrict__ mask,
                                                 const int* __restrict__ assign,
                                                 float* __restrict__ sums,
                                                 float* __restrict__ counts) {
    int wave = (int)((blockIdx.x * blockDim.x + threadIdx.x) >> 5);
    int lane = threadIdx.x & 31;
    if (wave >= NTOK) return;
    int br = wave / NN;
    int n  = wave % NN;
    if (mask[n] == 0) return;                        // masked-out: no contribution
    int b = n / PP, p = n % PP;
    const float* src = img + (((size_t)(br * BB + b)) * TT + (1 + p)) * DD;
    float x[24];
    float ss = 0.0f;
#pragma unroll
    for (int i = 0; i < 24; ++i) { x[i] = src[lane + 32 * i]; ss += x[i] * x[i]; }
    ss = waveSum(ss);
    float inv = 1.0f / fmaxf(sqrtf(ss), 1e-12f);
    int k = assign[wave];
    float* srow = sums + ((size_t)br * KK + k) * DD;
#pragma unroll
    for (int i = 0; i < 24; ++i) atomicAdd(&srow[lane + 32 * i], x[i] * inv);
    if (lane == 0) atomicAdd(&counts[br * KK + k], 1.0f);
}

// ---------------------------------------------------------------------------
// Kernel 4: mean / l2norm / EMA / l2norm / select. One wave per prototype.
// ---------------------------------------------------------------------------
__global__ void __launch_bounds__(256) k_finalize(const float* __restrict__ pro,
                                                  const float* __restrict__ sums,
                                                  const float* __restrict__ counts,
                                                  float* __restrict__ out) {
    int wave = (int)((blockIdx.x * blockDim.x + threadIdx.x) >> 5);
    int lane = threadIdx.x & 31;
    if (wave >= BR * KK) return;
    float cnt = counts[wave];
    const float* prow = pro + (size_t)wave * DD;
    const float* srow = sums + (size_t)wave * DD;
    float p[24], u[24];
    float denom = fmaxf(cnt, 1.0f);
    float ssm = 0.0f;
#pragma unroll
    for (int i = 0; i < 24; ++i) {
        p[i] = prow[lane + 32 * i];
        u[i] = srow[lane + 32 * i] / denom;          // mean
        ssm += u[i] * u[i];
    }
    ssm = waveSum(ssm);
    float invm = 1.0f / fmaxf(sqrtf(ssm), 1e-12f);
    float ssu = 0.0f;
#pragma unroll
    for (int i = 0; i < 24; ++i) {
        u[i] = 0.95f * p[i] + 0.05f * (u[i] * invm); // EMA
        ssu += u[i] * u[i];
    }
    ssu = waveSum(ssu);
    float invu = 1.0f / fmaxf(sqrtf(ssu), 1e-12f);
    float* orow = out + (size_t)wave * DD;
#pragma unroll
    for (int i = 0; i < 24; ++i)
        orow[lane + 32 * i] = (cnt > 0.0f) ? u[i] * invu : p[i];
}

// ---------------------------------------------------------------------------
// Host launcher
// ---------------------------------------------------------------------------
extern "C" void kernel_launch(void* const* d_in, const int* in_sizes, int n_in,
                              void* d_out, int out_size, void* d_ws, size_t ws_size,
                              hipStream_t stream) {
    (void)in_sizes; (void)n_in; (void)out_size; (void)ws_size;

    const float*         img  = (const float*)d_in[0];          // [4,32,577,768] f32
    const float*         pro  = (const float*)d_in[1];          // [4,2048,768]  f32
    const unsigned char* mask = (const unsigned char*)d_in[2];  // [32,576] bool
    float*               out  = (float*)d_out;                  // [4,2048,768]  f32

    // workspace layout (bytes)
    char* ws = (char*)d_ws;
    const size_t off_toksB  = 0;                                   // 113,246,208 B
    const size_t off_proB   = off_toksB + (size_t)NTOK * DD * 2;   //  12,582,912 B
    const size_t off_sums   = off_proB + (size_t)BR * KK * DD * 2; //  25,165,824 B
    const size_t off_counts = off_sums + (size_t)BR * KK * DD * 4; //      32,768 B
    const size_t off_assign = off_counts + (size_t)BR * KK * 4;    //     294,912 B

    __bf16* toksB  = (__bf16*)(ws + off_toksB);
    __bf16* proB   = (__bf16*)(ws + off_proB);
    float*  sums   = (float*)(ws + off_sums);
    float*  counts = (float*)(ws + off_counts);
    int*    assign = (int*)(ws + off_assign);

    // sums + counts are contiguous -> single zeroing pass (every call: graph-safe)
    size_t nzero = ((size_t)BR * KK * DD) + ((size_t)BR * KK);
    k_zero<<<2048, 256, 0, stream>>>(sums, nzero);

    k_norm_tokens<<<NTOK / 8, 256, 0, stream>>>(img, toksB);          // 9216 blocks
    k_cvt_protos<<<2048, 256, 0, stream>>>(pro, proB);
    k_argmax_wmma<<<(BR * (NN / 16)) / 8, 256, 2 * TILE_BYTES, stream>>>(toksB, proB, assign); // 576 blocks
    k_scatter<<<NTOK / 8, 256, 0, stream>>>(img, mask, assign, sums, counts);
    k_finalize<<<(BR * KK) / 8, 256, 0, stream>>>(pro, sums, counts, out);        // 1024 blocks
}
